// Head_14654428414298
// MI455X (gfx1250) — matrix operand order repaired and checked by reading, not compiled, not run on hardware
//
#include <hip/hip_runtime.h>

// MI455X (gfx1250) wave32 fused WMMA causal attention, one workgroup per batch.
// B=256 T=256 C=384 H=64.  HBM traffic ~117MB (x + out only): q/k/v live in LDS.
//   phase 0: stage f16-packed weights (WMMA B-operand layout) into LDS
//   phase 1: wave w computes q/k/v row-tile w via v_wmma_f32_16x16x32_f16 -> LDS
//   phase 2: wave w does causal attention for query tile w from LDS
// The weight-staging region is reused as per-wave softmax-weight scratch in
// phase 2 (regions are separated by the inter-phase barrier).

typedef __attribute__((ext_vector_type(16))) _Float16 v16h;
typedef __attribute__((ext_vector_type(8)))  _Float16 v8h;
typedef __attribute__((ext_vector_type(8)))  float    v8f;
typedef __attribute__((ext_vector_type(4)))  float    v4f;

#define B_DIM 256
#define T_DIM 256
#define C_DIM 384
#define H_DIM 64
#define NWAVE 16

// LDS layout, units = halves (_Float16)
#define QS 72                                  // q/k row stride (64 + 8 pad)
#define VS 264                                 // vt row stride (256 + 8 pad)
#define WSTR 272                               // wei row stride (256 + 16 pad)
#define OFF_Q  0
#define OFF_K  (OFF_Q + T_DIM * QS)            // 18432
#define OFF_VT (OFF_K + T_DIM * QS)            // 36864
#define OFF_WP (OFF_VT + H_DIM * VS)           // 53760 (wpack stage / wei scratch)
#define WPACK_HALVES (3 * 48 * 512)            // 73728 halves = 147456 B
#define SMEM_HALVES (OFF_WP + WPACK_HALVES)    // 127488 halves = 254976 B

static __device__ __forceinline__ v8f wmma_f16(v16h a, v16h b, v8f c) {
  // (neg_a, A, neg_b, B, c_mod, C, reuse_a, reuse_b)
  return __builtin_amdgcn_wmma_f32_16x16x32_f16(false, a, false, b, (short)0, c,
                                                false, false);
}

static __device__ __forceinline__ v8f zero8() {
  v8f z;
#pragma unroll
  for (int i = 0; i < 8; ++i) z[i] = 0.0f;
  return z;
}

// 16-halves B/A operand from two 16B-aligned chunks (LDS -> 2x ds_load_b128)
static __device__ __forceinline__ v16h ld16(const _Float16* p) {
  const v8h lo = *(const v8h*)p;
  const v8h hi = *(const v8h*)(p + 8);
  v16h r;
#pragma unroll
  for (int j = 0; j < 8; ++j) { r[j] = lo[j]; r[8 + j] = hi[j]; }
  return r;
}

// ---------------------------------------------------------------------------
// Kernel 0: pre-swizzle Wq/Wk/Wv ([C,H] fp32) into WMMA B-operand lane layout.
// B operand (32x16 f16, K=contraction): lane l holds column n=(l&15);
// lanes 0-15 hold K=0..7,16..23 (2/VGPR), lanes 16-31 hold K=8..15,24..31.
// wpack layout: [mat(3)][kstep(12)][ntile(4)][lane(32)][16 halves]
// ---------------------------------------------------------------------------
__global__ void pack_w_kernel(const float* __restrict__ Wq,
                              const float* __restrict__ Wk,
                              const float* __restrict__ Wv,
                              _Float16* __restrict__ wpack) {
  const int id   = blockIdx.x;          // 0..143 = mat*48 + ks*4 + nt
  const int mat  = id / 48;
  const int ks   = (id % 48) / 4;
  const int nt   = id % 4;
  const int lane = threadIdx.x;
  const int m    = lane & 15;
  const int hl   = lane >> 4;
  const float* W = (mat == 0) ? Wq : (mat == 1) ? Wk : Wv;
  const int n    = nt * 16 + m;
  _Float16* dst  = wpack + (size_t)id * 512 + lane * 16;
#pragma unroll
  for (int j = 0; j < 16; ++j) {
    const int kk = ks * 32 + (j < 8 ? 8 * hl + j : 16 + 8 * hl + (j - 8));
    dst[j] = (_Float16)W[kk * H_DIM + n];
  }
}

// ---------------------------------------------------------------------------
// Fused per-batch kernel: 16 waves (512 threads), one workgroup per batch.
// ---------------------------------------------------------------------------
__global__ __launch_bounds__(NWAVE * 32, 1)
void fused_attn_kernel(const float* __restrict__ x,
                       const _Float16* __restrict__ wpack,
                       float* __restrict__ out) {
  __shared__ __align__(32) _Float16 smem[SMEM_HALVES];   // ~249 KB of WGP LDS

  const int b    = blockIdx.x;          // batch
  const int tid  = threadIdx.x;
  const int w    = tid >> 5;            // wave id = row tile = query tile
  const int lane = tid & 31;
  const int m    = lane & 15;
  const int hl   = lane >> 4;

  // ---- phase 0: cooperative stage of packed weights into LDS ----
  {
    const uint4* src = (const uint4*)wpack;
    uint4* dst = (uint4*)(smem + OFF_WP);
#pragma unroll
    for (int i = 0; i < (WPACK_HALVES * 2 / 16) / (NWAVE * 32); ++i)   // 18 iters
      dst[tid + i * NWAVE * 32] = src[tid + i * NWAVE * 32];
  }
  __syncthreads();

  // ---- phase 1: q/k/v row-tile w -> LDS ----
  // A operand (16x32 f16): lane l = row m; K-locals 8*hl+0..7 in VGPR0-3,
  // 16+8*hl+0..7 in VGPR4-7 (ISA 7.12.2).
  {
    const float* xr = x + ((size_t)b * T_DIM + w * 16 + m) * C_DIM;
    v8f accQ[4], accK[4], accV[4];
#pragma unroll
    for (int nt = 0; nt < 4; ++nt) {
      accQ[nt] = zero8(); accK[nt] = zero8(); accV[nt] = zero8();
    }

#pragma unroll
    for (int ks = 0; ks < 12; ++ks) {
      const int kk = ks * 32;
      const float* p0 = xr + kk + 8 * hl;
      const float* p1 = xr + kk + 16 + 8 * hl;
      const v4f c00 = *(const v4f*)(p0);
      const v4f c01 = *(const v4f*)(p0 + 4);
      const v4f c10 = *(const v4f*)(p1);
      const v4f c11 = *(const v4f*)(p1 + 4);
      v16h a;
#pragma unroll
      for (int j = 0; j < 4; ++j) {
        a[j]      = (_Float16)c00[j];
        a[4 + j]  = (_Float16)c01[j];
        a[8 + j]  = (_Float16)c10[j];
        a[12 + j] = (_Float16)c11[j];
      }
      const _Float16* wp = smem + OFF_WP + (size_t)(ks * 4) * 512 + lane * 16;
#pragma unroll
      for (int nt = 0; nt < 4; ++nt) {
        const v16h bq = ld16(wp + nt * 512);              // Wq tiles
        const v16h bk = ld16(wp + 24576 + nt * 512);      // Wk tiles
        const v16h bv = ld16(wp + 49152 + nt * 512);      // Wv tiles
        accQ[nt] = wmma_f16(a, bq, accQ[nt]);
        accK[nt] = wmma_f16(a, bk, accK[nt]);
        accV[nt] = wmma_f16(a, bv, accV[nt]);
      }
    }

    // C/D layout: VGPR r holds (M=r+8*hl, N=l&15)
    const int t0 = w * 16;
#pragma unroll
    for (int nt = 0; nt < 4; ++nt) {
#pragma unroll
      for (int r = 0; r < 8; ++r) {
        const int tt = t0 + r + 8 * hl;
        const int h  = nt * 16 + m;
        smem[OFF_Q + tt * QS + h]  = (_Float16)accQ[nt][r];
        smem[OFF_K + tt * QS + h]  = (_Float16)accK[nt][r];
        smem[OFF_VT + h * VS + tt] = (_Float16)accV[nt][r];   // v transposed
      }
    }
  }
  __syncthreads();   // also separates wpack reads from wei-scratch writes

  // ---- phase 2: causal attention for query tile qt = w ----
  const int qt = w;
  _Float16* mywei = smem + OFF_WP + w * (16 * WSTR);   // per-wave scratch

  // Q A-operands (two K-chunks over H=64)
  const _Float16* qrow = smem + OFF_Q + (qt * 16 + m) * QS;
  v16h aq0, aq1;
  {
    const v8h a00 = *(const v8h*)(qrow + 16 * hl);
    const v8h a01 = *(const v8h*)(qrow + 16 * hl + 8);
    const v8h a10 = *(const v8h*)(qrow + 32 + 16 * hl);
    const v8h a11 = *(const v8h*)(qrow + 32 + 16 * hl + 8);
#pragma unroll
    for (int j = 0; j < 8; ++j) {
      aq0[j] = a00[j]; aq0[8 + j] = a01[j];
      aq1[j] = a10[j]; aq1[8 + j] = a11[j];
    }
  }

  // scores: only causal tiles n <= qt (wave-uniform branch, EXEC stays full)
  v8f accW[16];
#pragma unroll
  for (int n = 0; n < 16; ++n) accW[n] = zero8();
#pragma unroll
  for (int n = 0; n < 16; ++n) {
    if (n <= qt) {
      const _Float16* kr = smem + OFF_K + (n * 16 + m) * QS + 16 * hl;
      accW[n] = wmma_f16(aq0, ld16(kr), accW[n]);
      accW[n] = wmma_f16(aq1, ld16(kr + 32), accW[n]);
    }
  }

  // scale + causal mask + row max
  const float scale = 0.05103103630798288f;   // 384^-0.5 (C, per reference)
  float rmax[8];
#pragma unroll
  for (int r = 0; r < 8; ++r) rmax[r] = -__builtin_inff();
#pragma unroll
  for (int n = 0; n < 16; ++n) {
    if (n <= qt) {
#pragma unroll
      for (int r = 0; r < 8; ++r) {
        float v = accW[n][r] * scale;
        const int qi = r + 8 * hl;
        if (n == qt && m > qi) v = -__builtin_inff();   // key > query
        accW[n][r] = v;
        rmax[r] = fmaxf(rmax[r], v);
      }
    }
  }
#pragma unroll
  for (int r = 0; r < 8; ++r) {
#pragma unroll
    for (int off = 1; off < 16; off <<= 1)
      rmax[r] = fmaxf(rmax[r], __shfl_xor(rmax[r], off, 16));
  }

  // exp + row sum
  float rsum[8];
#pragma unroll
  for (int r = 0; r < 8; ++r) rsum[r] = 0.0f;
#pragma unroll
  for (int n = 0; n < 16; ++n) {
    if (n <= qt) {
#pragma unroll
      for (int r = 0; r < 8; ++r) {
        const float e = __expf(accW[n][r] - rmax[r]);
        accW[n][r] = e;
        rsum[r] += e;
      }
    }
  }
#pragma unroll
  for (int r = 0; r < 8; ++r) {
#pragma unroll
    for (int off = 1; off < 16; off <<= 1)
      rsum[r] += __shfl_xor(rsum[r], off, 16);
  }
  float rinv[8];
#pragma unroll
  for (int r = 0; r < 8; ++r) rinv[r] = 1.0f / rsum[r];

  // normalized weights -> per-wave LDS scratch (C-layout -> A-operand reshape)
#pragma unroll
  for (int n = 0; n < 16; ++n) {
    if (n <= qt) {
#pragma unroll
      for (int r = 0; r < 8; ++r)
        mywei[(r + 8 * hl) * WSTR + n * 16 + m] =
            (_Float16)(accW[n][r] * rinv[r]);
    }
  }
  if ((qt & 1) == 0) {   // zero the dangling half of the last 32-key chunk
#pragma unroll
    for (int r = 0; r < 8; ++r)
      mywei[(r + 8 * hl) * WSTR + (qt + 1) * 16 + m] = (_Float16)0.0f;
  }
  __syncthreads();   // orders per-wave LDS scratch (cross-lane) + cheap

  // out = wei @ V  (Vt rows give contiguous B-operand columns from LDS)
  v8f acc2[4];
#pragma unroll
  for (int nt = 0; nt < 4; ++nt) acc2[nt] = zero8();
  const int nchunks = (qt >> 1) + 1;
#pragma unroll
  for (int cc = 0; cc < 8; ++cc) {
    if (cc < nchunks) {
      const _Float16* lrow = mywei + m * WSTR + cc * 32;
      const v8h w0 = *(const v8h*)(lrow + 16 * hl);
      const v8h w1 = *(const v8h*)(lrow + 16 * hl + 8);
      v16h aw;
#pragma unroll
      for (int j = 0; j < 8; ++j) { aw[j] = w0[j]; aw[8 + j] = w1[j]; }
#pragma unroll
      for (int nt = 0; nt < 4; ++nt) {
        const _Float16* vr =
            smem + OFF_VT + (nt * 16 + m) * VS + cc * 32 + 16 * hl;
        acc2[nt] = wmma_f16(aw, ld16(vr), acc2[nt]);
      }
    }
  }

#pragma unroll
  for (int nt = 0; nt < 4; ++nt) {
#pragma unroll
    for (int r = 0; r < 8; ++r)
      out[((size_t)b * T_DIM + qt * 16 + r + 8 * hl) * H_DIM + nt * 16 + m] =
          acc2[nt][r];
  }
}

// ---------------------------------------------------------------------------
extern "C" void kernel_launch(void* const* d_in, const int* in_sizes, int n_in,
                              void* d_out, int out_size, void* d_ws, size_t ws_size,
                              hipStream_t stream) {
  (void)in_sizes; (void)n_in; (void)out_size; (void)ws_size;
  const float* x  = (const float*)d_in[0];
  const float* Wq = (const float*)d_in[1];
  const float* Wk = (const float*)d_in[2];
  const float* Wv = (const float*)d_in[3];
  float* out = (float*)d_out;

  _Float16* wpack = (_Float16*)d_ws;   // 147,456 bytes of scratch

  pack_w_kernel<<<144, 32, 0, stream>>>(Wq, Wk, Wv, wpack);
  fused_attn_kernel<<<B_DIM, NWAVE * 32, 0, stream>>>(x, wpack, out);
}
